// MambaExpert_9440338116673
// MI455X (gfx1250) — matrix operand order repaired and checked by reading, not compiled, not run on hardware
//
#include <hip/hip_runtime.h>

// Mamba block forward for MI455X (gfx1250, wave32, WMMA, async global->LDS).
// All GEMM operands are bf16 in global memory (pre-converted / produced bf16 by
// epilogues) so LDS tiles are raw byte copies -> GLOBAL_LOAD_ASYNC_TO_LDS_B128.

#define B_SZ     8
#define SEQ_L    2048
#define DM       768
#define DI       1536
#define DS       16
#define DTR      192
#define MTOK     (B_SZ * SEQ_L)      // 16384 tokens
#define NSEG     16
#define CHUNK    (SEQ_L / NSEG)      // 128

typedef __attribute__((ext_vector_type(16))) __bf16 v16bf;
typedef __attribute__((ext_vector_type(8)))  __bf16 v8bf;
typedef __attribute__((ext_vector_type(4)))  __bf16 v4bf;
typedef __attribute__((ext_vector_type(8)))  float  v8f;
typedef __attribute__((ext_vector_type(4)))  int    v4i;

#if defined(__gfx1250__) && __has_builtin(__builtin_amdgcn_global_load_async_to_lds_b128)
#define HAVE_ASYNC_LDS 1
#endif

enum { EPI_SILU_BF = 0, EPI_SPLIT_DBC = 1, EPI_SOFTPLUS_F = 2, EPI_F32 = 3 };

__device__ __forceinline__ float siluf(float v) {
    return v * (1.0f / (1.0f + __expf(-v)));
}
__device__ __forceinline__ float softplus_f(float v) {
    return (v > 20.0f) ? v : log1pf(__expf(v));
}
__device__ __forceinline__ v16bf cat16(v8bf lo, v8bf hi) {
    union { v16bf v; v8bf h[2]; } u;
    u.h[0] = lo; u.h[1] = hi;
    return u.v;
}

// 16-byte global -> LDS copy; async (ASYNCcnt-tracked) when available.
__device__ __forceinline__ void cp16_g2l(const __bf16* g, __bf16* l) {
#if defined(HAVE_ASYNC_LDS)
    __builtin_amdgcn_global_load_async_to_lds_b128(
        (__attribute__((address_space(1))) v4i*)(void*)(const void*)g,
        (__attribute__((address_space(3))) v4i*)(void*)l, 0, 0);
#else
    *(int4*)l = *(const int4*)g;
#endif
}
__device__ __forceinline__ void cp_async_wait0() {
#if defined(HAVE_ASYNC_LDS)
#if __has_builtin(__builtin_amdgcn_s_wait_asynccnt)
    __builtin_amdgcn_s_wait_asynccnt(0);
#else
    asm volatile("s_wait_asynccnt 0x0" ::: "memory");
#endif
#endif
}

// f32 -> bf16 bulk convert (n must be a multiple of 4)
__global__ __launch_bounds__(256) void cvt_bf16(
    const float* __restrict__ s, __bf16* __restrict__ d, int n4)
{
    const int i = blockIdx.x * 256 + threadIdx.x;
    if (i < n4) {
        float4 f = *(const float4*)(s + (size_t)i * 4);
        v4bf b = { (__bf16)f.x, (__bf16)f.y, (__bf16)f.z, (__bf16)f.w };
        *(v4bf*)(d + (size_t)i * 4) = b;
    }
}

// ---------------------------------------------------------------------------
// bf16 WMMA GEMM: out[m][n] = epi( sum_k A[m][k] * W[n][k] ), A:[M,LDA] W:[N,LDW]
// both bf16 row-major; K multiple of 32; tiles divide all dims exactly.
// ---------------------------------------------------------------------------
template <int BM, int BN, int WAVES_M, int WAVES_N, int LDA, int LDW, int EPI>
__global__ __launch_bounds__(256) void wmma_gemm(
    const __bf16* __restrict__ A, const __bf16* __restrict__ W,
    float* __restrict__ OutF, __bf16* __restrict__ OutB, int ldo,
    int K, const float* __restrict__ bias)
{
    constexpr int WTM = BM / WAVES_M;
    constexpr int WTN = BN / WAVES_N;
    constexpr int TM  = WTM / 16;
    constexpr int TN  = WTN / 16;

    __shared__ __align__(16) __bf16 Asm[BM * 32];
    __shared__ __align__(16) __bf16 Bsm[BN * 32];

    const int tid  = threadIdx.x;
    const int lane = tid & 31;
    const int wave = tid >> 5;
    const int lrow = lane & 15;
    const int lhi  = lane >> 4;
    const int m0 = blockIdx.y * BM;
    const int n0 = blockIdx.x * BN;
    const int wm = (wave % WAVES_M) * WTM;
    const int wn = (wave / WAVES_M) * WTN;

    v8f acc[TM][TN] = {};

    for (int k0 = 0; k0 < K; k0 += 32) {
        __syncthreads();               // previous-iteration LDS reads done
        // Stage A (BM x 32 bf16) and B (BN x 32 bf16): raw 16B async copies.
        for (int i = tid; i < BM * 4; i += 256) {
            const int r = i >> 2, c = (i & 3) << 3;
            cp16_g2l(A + (size_t)(m0 + r) * LDA + k0 + c, Asm + r * 32 + c);
        }
        for (int i = tid; i < BN * 4; i += 256) {
            const int r = i >> 2, c = (i & 3) << 3;
            cp16_g2l(W + (size_t)(n0 + r) * LDW + k0 + c, Bsm + r * 32 + c);
        }
        cp_async_wait0();
        __syncthreads();

        // Fragments per ISA 7.12.2 (wave32):
        // A 16x32: lanes<16 K {0..7,16..23}; lanes>=16 K {8..15,24..31}
        // B 32x16: lanes<16 K {0..7,16..23} of col lrow; lanes>=16 K {8..15,24..31}
        v16bf afrag[TM], bfrag[TN];
#pragma unroll
        for (int i = 0; i < TM; ++i) {
            const __bf16* p = Asm + (wm + i * 16 + lrow) * 32 + lhi * 8;
            afrag[i] = cat16(*(const v8bf*)p, *(const v8bf*)(p + 16));
        }
#pragma unroll
        for (int j = 0; j < TN; ++j) {
            const __bf16* p = Bsm + (wn + j * 16 + lrow) * 32 + lhi * 8;
            bfrag[j] = cat16(*(const v8bf*)p, *(const v8bf*)(p + 16));
        }
#pragma unroll
        for (int i = 0; i < TM; ++i)
#pragma unroll
            for (int j = 0; j < TN; ++j)
                acc[i][j] = __builtin_amdgcn_wmma_f32_16x16x32_bf16(
                    false, afrag[i], false, bfrag[j],
                    (short)0, acc[i][j], false, false);
    }

    // Epilogue. C/D layout: VGPR r -> row (r + 8*lhi), col lrow.
#pragma unroll
    for (int i = 0; i < TM; ++i)
#pragma unroll
        for (int j = 0; j < TN; ++j) {
            const int nn = n0 + wn + j * 16 + lrow;
            float bv = 0.0f;
            if (EPI == EPI_SOFTPLUS_F) bv = bias[nn];
#pragma unroll
            for (int r = 0; r < 8; ++r) {
                const int mm = m0 + wm + i * 16 + lhi * 8 + r;
                const float v = acc[i][j][r];
                if (EPI == EPI_SILU_BF) {
                    OutB[(size_t)mm * ldo + nn] = (__bf16)siluf(v);
                } else if (EPI == EPI_SPLIT_DBC) {
                    if (nn < DTR) OutB[(size_t)mm * DTR + nn] = (__bf16)v;
                    else          OutF[(size_t)mm * (2 * DS) + (nn - DTR)] = v;
                } else if (EPI == EPI_SOFTPLUS_F) {
                    OutF[(size_t)mm * ldo + nn] = softplus_f(v + bv);
                } else {
                    OutF[(size_t)mm * ldo + nn] = v;
                }
            }
        }
}

// ---------------------------------------------------------------------------
// Chunked selective scan (diagonal-linear recurrence composes per segment:
// h_out = P*h_in + hL, P = prod(dA)). 3 phases -> NSEG-fold parallelism.
// ---------------------------------------------------------------------------
__global__ __launch_bounds__(256) void scan_part1(
    const __bf16* __restrict__ xz,   // [MTOK, 2*DI] silu(x)|silu(z) bf16
    const float*  __restrict__ bc,   // [MTOK, 32]   B|C f32
    const float*  __restrict__ delta,// [MTOK, DI]   f32
    const float*  __restrict__ A_log,// [DI, DS]
    float* __restrict__ segP, float* __restrict__ segH)
{
    const int gid = blockIdx.x * 256 + threadIdx.x;   // (b, seg, d)
    const int d   = gid % DI;
    const int bs  = gid / DI;
    const int seg = bs % NSEG;
    const int b   = bs / NSEG;

    float a[DS], P[DS], h[DS];
#pragma unroll
    for (int n = 0; n < DS; ++n) {
        a[n] = -__expf(A_log[d * DS + n]);
        P[n] = 1.0f; h[n] = 0.0f;
    }

    const int t0 = seg * CHUNK;
    for (int t = 0; t < CHUNK; ++t) {
        const size_t m = (size_t)b * SEQ_L + t0 + t;
        const float x_t = (float)xz[m * (2 * DI) + d];
        const float dt  = delta[m * DI + d];
        const float* bp = bc + m * (2 * DS);
        float Bm[DS];
        *(float4*)(Bm + 0)  = *(const float4*)(bp + 0);
        *(float4*)(Bm + 4)  = *(const float4*)(bp + 4);
        *(float4*)(Bm + 8)  = *(const float4*)(bp + 8);
        *(float4*)(Bm + 12) = *(const float4*)(bp + 12);
        const float dx = dt * x_t;
#pragma unroll
        for (int n = 0; n < DS; ++n) {
            const float dA = __expf(dt * a[n]);
            P[n] *= dA;
            h[n] = dA * h[n] + dx * Bm[n];
        }
    }
    const size_t base = ((size_t)(b * NSEG + seg) * DS) * DI + d;
#pragma unroll
    for (int n = 0; n < DS; ++n) {
        segP[base + (size_t)n * DI] = P[n];
        segH[base + (size_t)n * DI] = h[n];
    }
}

__global__ __launch_bounds__(256) void scan_combine(
    const float* __restrict__ segP, const float* __restrict__ segH,
    float* __restrict__ hIn)
{
    const int gid = blockIdx.x * 256 + threadIdx.x;   // (b, d)
    const int d = gid % DI;
    const int b = gid / DI;
    float h[DS];
#pragma unroll
    for (int n = 0; n < DS; ++n) h[n] = 0.0f;
    for (int seg = 0; seg < NSEG; ++seg) {
        const size_t base = ((size_t)(b * NSEG + seg) * DS) * DI + d;
#pragma unroll
        for (int n = 0; n < DS; ++n) {
            const size_t idx = base + (size_t)n * DI;
            hIn[idx] = h[n];
            h[n] = segP[idx] * h[n] + segH[idx];
        }
    }
}

__global__ __launch_bounds__(256) void scan_part2(
    const __bf16* __restrict__ xz, const float* __restrict__ bc,
    const float* __restrict__ delta, const float* __restrict__ A_log,
    const float* __restrict__ Dvec, const float* __restrict__ hIn,
    __bf16* __restrict__ yg)
{
    const int gid = blockIdx.x * 256 + threadIdx.x;   // (b, seg, d)
    const int d   = gid % DI;
    const int bs  = gid / DI;
    const int seg = bs % NSEG;
    const int b   = bs / NSEG;

    float a[DS], h[DS];
    const size_t base = ((size_t)(b * NSEG + seg) * DS) * DI + d;
#pragma unroll
    for (int n = 0; n < DS; ++n) {
        a[n] = -__expf(A_log[d * DS + n]);
        h[n] = hIn[base + (size_t)n * DI];
    }
    const float Dd = Dvec[d];

    const int t0 = seg * CHUNK;
    for (int t = 0; t < CHUNK; ++t) {
        const size_t m = (size_t)b * SEQ_L + t0 + t;
        const float x_t = (float)xz[m * (2 * DI) + d];
        const float z_s = (float)xz[m * (2 * DI) + DI + d];
        const float dt  = delta[m * DI + d];
        const float* bp = bc + m * (2 * DS);
        float Bm[DS], Cm[DS];
        *(float4*)(Bm + 0)  = *(const float4*)(bp + 0);
        *(float4*)(Bm + 4)  = *(const float4*)(bp + 4);
        *(float4*)(Bm + 8)  = *(const float4*)(bp + 8);
        *(float4*)(Bm + 12) = *(const float4*)(bp + 12);
        *(float4*)(Cm + 0)  = *(const float4*)(bp + 16);
        *(float4*)(Cm + 4)  = *(const float4*)(bp + 20);
        *(float4*)(Cm + 8)  = *(const float4*)(bp + 24);
        *(float4*)(Cm + 12) = *(const float4*)(bp + 28);
        const float dx = dt * x_t;
        float y = 0.0f;
#pragma unroll
        for (int n = 0; n < DS; ++n) {
            const float dA = __expf(dt * a[n]);
            h[n] = dA * h[n] + dx * Bm[n];
            y += h[n] * Cm[n];
        }
        yg[m * DI + d] = (__bf16)((y + x_t * Dd) * z_s);
    }
}

// ---------------------------------------------------------------------------
extern "C" void kernel_launch(void* const* d_in, const int* in_sizes, int n_in,
                              void* d_out, int out_size, void* d_ws, size_t ws_size,
                              hipStream_t stream) {
    const float* u     = (const float*)d_in[0];  // [8,2048,768]
    const float* W_in  = (const float*)d_in[1];  // [3072,768]
    const float* W_x   = (const float*)d_in[2];  // [224,1536]
    const float* W_dt  = (const float*)d_in[3];  // [1536,192]
    const float* b_dt  = (const float*)d_in[4];  // [1536]
    const float* A_log = (const float*)d_in[5];  // [1536,16]
    const float* Dvec  = (const float*)d_in[6];  // [1536]
    const float* W_out = (const float*)d_in[7];  // [768,1536]
    float* out = (float*)d_out;                  // [8,2048,768]

    char* p = (char*)d_ws;
    auto alloc = [&](size_t bytes) {
        char* r = p; p += (bytes + 255) & ~(size_t)255; return r;
    };
    __bf16* u_bf    = (__bf16*)alloc((size_t)MTOK * DM * 2);
    __bf16* Win_bf  = (__bf16*)alloc((size_t)(2 * DI) * DM * 2);
    __bf16* Wx_bf   = (__bf16*)alloc((size_t)(DTR + 2 * DS) * DI * 2);
    __bf16* Wdt_bf  = (__bf16*)alloc((size_t)DI * DTR * 2);
    __bf16* Wout_bf = (__bf16*)alloc((size_t)DM * DI * 2);
    __bf16* xz_bf   = (__bf16*)alloc((size_t)MTOK * (2 * DI) * 2);
    __bf16* dtb_bf  = (__bf16*)alloc((size_t)MTOK * DTR * 2);
    float*  bc      = (float*) alloc((size_t)MTOK * (2 * DS) * 4);
    float*  delta   = (float*) alloc((size_t)MTOK * DI * 4);
    __bf16* yg_bf   = (__bf16*)alloc((size_t)MTOK * DI * 2);
    float*  segP    = (float*) alloc((size_t)B_SZ * NSEG * DS * DI * 4);
    float*  segH    = (float*) alloc((size_t)B_SZ * NSEG * DS * DI * 4);
    float*  hIn     = (float*) alloc((size_t)B_SZ * NSEG * DS * DI * 4);

    // 0) one-time f32->bf16 conversion of u and all weights
    auto cvt = [&](const float* s, __bf16* d, size_t n) {
        int n4 = (int)(n / 4);
        cvt_bf16<<<(n4 + 255) / 256, 256, 0, stream>>>(s, d, n4);
    };
    cvt(u,     u_bf,    (size_t)MTOK * DM);
    cvt(W_in,  Win_bf,  (size_t)(2 * DI) * DM);
    cvt(W_x,   Wx_bf,   (size_t)(DTR + 2 * DS) * DI);
    cvt(W_dt,  Wdt_bf,  (size_t)DI * DTR);
    cvt(W_out, Wout_bf, (size_t)DM * DI);

    // 1) in_proj + SiLU (both halves): xz_bf = silu(u @ W_in^T)
    wmma_gemm<128, 64, 4, 2, DM, DM, EPI_SILU_BF>
        <<<dim3((2 * DI) / 64, MTOK / 128), 256, 0, stream>>>(
        u_bf, Win_bf, nullptr, xz_bf, 2 * DI, DM, nullptr);
    // 2) x_proj (N=224): dt part -> bf16, B|C part -> f32
    wmma_gemm<128, 32, 8, 1, 2 * DI, DI, EPI_SPLIT_DBC>
        <<<dim3((DTR + 2 * DS) / 32, MTOK / 128), 256, 0, stream>>>(
        xz_bf, Wx_bf, bc, dtb_bf, 0, DI, nullptr);
    // 3) dt_proj: delta = softplus(dt @ W_dt^T + b_dt), f32 for the scan
    wmma_gemm<128, 64, 4, 2, DTR, DTR, EPI_SOFTPLUS_F>
        <<<dim3(DI / 64, MTOK / 128), 256, 0, stream>>>(
        dtb_bf, Wdt_bf, delta, nullptr, DI, DTR, b_dt);
    // 4) chunked selective scan
    scan_part1<<<(B_SZ * NSEG * DI) / 256, 256, 0, stream>>>(
        xz_bf, bc, delta, A_log, segP, segH);
    scan_combine<<<(B_SZ * DI) / 256, 256, 0, stream>>>(segP, segH, hIn);
    scan_part2<<<(B_SZ * NSEG * DI) / 256, 256, 0, stream>>>(
        xz_bf, bc, delta, A_log, Dvec, hIn, yg_bf);
    // 5) out_proj: out = yg @ W_out^T (f32 out)
    wmma_gemm<128, 64, 4, 2, DI, DI, EPI_F32>
        <<<dim3(DM / 64, MTOK / 128), 256, 0, stream>>>(
        yg_bf, Wout_bf, out, nullptr, DM, DI, nullptr);
}
